// Seq2Seq_66159676227873
// MI455X (gfx1250) — matrix operand order repaired and checked by reading, not compiled, not run on hardware
//
#include <hip/hip_runtime.h>
#include <hip/hip_bf16.h>
#include <stdint.h>
#include <stddef.h>

// ---------------------------------------------------------------------------
// Seq2Seq (2-layer LSTM encoder + autoregressive 2-layer LSTM decoder) for
// MI455X / gfx1250.  All matmuls run on v_wmma_f32_16x16x32_bf16 (wave32).
// Pregates stored in WMMA-C tiled layout; recurrence double-buffers the next
// step's pregates into LDS via global_load_async_to_lds (ASYNCcnt).
// ---------------------------------------------------------------------------

typedef __bf16 bf16_t;
typedef __attribute__((ext_vector_type(16))) __bf16 bf16x16;
typedef __attribute__((ext_vector_type(8)))  __bf16 bf16x8;
typedef __attribute__((ext_vector_type(8)))  float  f32x8;
typedef int v4i_t __attribute__((vector_size(16)));   // matches builtin param

#define HD     128
#define FOURH  512
#define OUTD   88
#define IND    1025
#define BT     16        // batch == WMMA M tile
#define SRCLEN 2048
#define TRGLEN 2048
#define KC0    33        // ceil(1025/32)

#if defined(__has_builtin)
#if __has_builtin(__builtin_amdgcn_global_load_async_to_lds_b128)
#define HAVE_ASYNC 1
#endif
#if __has_builtin(__builtin_amdgcn_s_wait_asynccnt)
#define HAVE_ASYNC_WAIT 1
#endif
#endif
#ifndef HAVE_ASYNC
#define HAVE_ASYNC 0
#endif
#ifndef HAVE_ASYNC_WAIT
#define HAVE_ASYNC_WAIT 0
#endif

#define AS1 __attribute__((address_space(1)))
#define AS3 __attribute__((address_space(3)))

static __device__ __forceinline__ bf16_t f2bf(float f) {
  unsigned u = __builtin_bit_cast(unsigned, f);
  unsigned r = u + 0x7FFFu + ((u >> 16) & 1u);   // round-to-nearest-even
  unsigned short s = (unsigned short)(r >> 16);
  return __builtin_bit_cast(bf16_t, s);
}

static __device__ __forceinline__ float sigm(float x) {
  return __builtin_amdgcn_rcpf(1.0f + __expf(-x));
}
static __device__ __forceinline__ float tanh_(float x) {
  return 2.0f * sigm(2.0f * x) - 1.0f;
}

#if HAVE_ASYNC
static __device__ __forceinline__ void async_g2l_b128(const void* g, void* l) {
  __builtin_amdgcn_global_load_async_to_lds_b128(
      (AS1 v4i_t*)(uintptr_t)g,
      (AS3 v4i_t*)(uint32_t)(uintptr_t)l, 0, 0);
}
static __device__ __forceinline__ void wait_async_le8() {
#if HAVE_ASYNC_WAIT
  __builtin_amdgcn_s_wait_asynccnt(8);
#else
  asm volatile("s_wait_asynccnt 0x8" ::: "memory");
#endif
}
#endif

// A fragment (16x32 bf16, M=batch row per lane) from an LDS tile of row
// stride `strideK` bf16 elements.  Lane L<16: K = kc*32 + {0..7,16..23};
// lanes 16..31: K = kc*32 + {8..15,24..31}.  Two 16B loads per lane.
static __device__ __forceinline__ bf16x16 load_afrag(const bf16_t* lds,
                                                     int strideK, int kc) {
  int lane = threadIdx.x & 31;
  int m  = lane & 15;
  int hi = lane >> 4;
  const bf16_t* p = lds + m * strideK + kc * 32 + hi * 8;
  bf16x8 lo = *(const bf16x8*)(p);
  bf16x8 hh = *(const bf16x8*)(p + 16);
  return __builtin_shufflevector(lo, hh, 0,1,2,3,4,5,6,7,8,9,10,11,12,13,14,15);
}

// B fragment (32x16 bf16, N column per lane) from a pre-transposed weight
// matrix WT[K][ncols] (bf16, row-major).  Lane L<16: K = kc*32 + {0..7,16..23};
// lanes 16..31: K = kc*32 + {8..15,24..31}.  Strided scalar loads.
static __device__ __forceinline__ bf16x16 load_bfrag(const bf16_t* WT,
                                                     int ncols, int kc, int nn) {
  int lane = threadIdx.x & 31;
  int hi = lane >> 4;
  const bf16_t* p = WT + (size_t)(kc * 32 + hi * 8) * ncols + nn + (lane & 15);
  bf16x16 v;
#pragma unroll
  for (int s = 0; s < 8; s++) {
    v[s] = p[(size_t)s * ncols];
    v[s + 8] = p[(size_t)(s + 16) * ncols];
  }
  return v;
}

static __device__ __forceinline__ f32x8 wmma_bf16(bf16x16 a, bf16x16 b, f32x8 c) {
  return __builtin_amdgcn_wmma_f32_16x16x32_bf16(false, a, false, b,
                                                 (short)0, c, false, false);
}

// ---------------------------------------------------------------------------
// K0: pack all weights into bf16 B-operand layout (transposed, zero padded)
// and combine biases.
// ---------------------------------------------------------------------------
__global__ __launch_bounds__(256) void k_prep(
    const float* eWih0, const float* eWhh0, const float* ebih0, const float* ebhh0,
    const float* eWih1, const float* eWhh1, const float* ebih1, const float* ebhh1,
    const float* dWih0, const float* dWhh0, const float* dbih0, const float* dbhh0,
    const float* dWih1, const float* dWhh1, const float* dbih1, const float* dbhh1,
    const float* fcW, const float* fcb,
    bf16_t* WTih0, bf16_t* WThh0, bf16_t* WTih1, bf16_t* WThh1,
    bf16_t* WTd0, bf16_t* WTd1, bf16_t* WTfc,
    float* be0, float* be1, float* bd0, float* bd1, float* bfc) {
  int tid = blockIdx.x * blockDim.x + threadIdx.x;
  int nth = gridDim.x * blockDim.x;
  const bf16_t z = f2bf(0.0f);

  for (int i = tid; i < KC0 * 32 * FOURH; i += nth) {       // [1056][512]
    int k = i / FOURH, n = i % FOURH;
    WTih0[i] = (k < IND) ? f2bf(eWih0[(size_t)n * IND + k]) : z;
  }
  for (int i = tid; i < HD * FOURH; i += nth) {             // [128][512] x3
    int k = i / FOURH, n = i % FOURH;
    WThh0[i] = f2bf(eWhh0[(size_t)n * HD + k]);
    WTih1[i] = f2bf(eWih1[(size_t)n * HD + k]);
    WThh1[i] = f2bf(eWhh1[(size_t)n * HD + k]);
  }
  for (int i = tid; i < 224 * FOURH; i += nth) {            // [96+128][512]
    int k = i / FOURH, n = i % FOURH;
    bf16_t v;
    if (k < 96) v = (k < OUTD) ? f2bf(dWih0[(size_t)n * OUTD + k]) : z;
    else        v = f2bf(dWhh0[(size_t)n * HD + (k - 96)]);
    WTd0[i] = v;
  }
  for (int i = tid; i < 256 * FOURH; i += nth) {            // [128+128][512]
    int k = i / FOURH, n = i % FOURH;
    WTd1[i] = (k < HD) ? f2bf(dWih1[(size_t)n * HD + k])
                       : f2bf(dWhh1[(size_t)n * HD + (k - HD)]);
  }
  for (int i = tid; i < HD * 96; i += nth) {                // [128][96]
    int k = i / 96, n = i % 96;
    WTfc[i] = (n < OUTD) ? f2bf(fcW[(size_t)n * HD + k]) : z;
  }
  for (int n = tid; n < FOURH; n += nth) {
    be0[n] = ebih0[n] + ebhh0[n];
    be1[n] = ebih1[n] + ebhh1[n];
    bd0[n] = dbih0[n] + dbhh0[n];
    bd1[n] = dbih1[n] + dbhh1[n];
  }
  for (int n = tid; n < 96; n += nth) bfc[n] = (n < OUTD) ? fcb[n] : 0.0f;
}

// ---------------------------------------------------------------------------
// K1: batched input-to-hidden pregates, written in WMMA-C tiled layout:
//   G[((t*32 + tile)*32 + lane)*8 + e]   (tile = n/16, 8 f32 per lane).
// One workgroup per timestep t (M=16 rows = batch), 8 waves x 4 N-tiles.
// Software-pipelined: next A/B fragments load while current WMMAs execute.
// ---------------------------------------------------------------------------
__global__ __launch_bounds__(256) void k_pregate(
    const float* __restrict__ A, long strideB, long strideT,
    int Kvalid, int KC, const bf16_t* __restrict__ WT,
    const float* __restrict__ bias, float* __restrict__ G) {
  __shared__ __align__(64) bf16_t Ald[BT * KC0 * 32];
  const int t = blockIdx.x;
  const int tid = threadIdx.x;
  const int strideK = KC * 32;

  for (int idx = tid; idx < BT * strideK; idx += 256) {
    int m = idx / strideK, k = idx - m * strideK;
    float v = (k < Kvalid) ? A[(long)m * strideB + (long)t * strideT + k] : 0.0f;
    Ald[idx] = f2bf(v);
  }
  __syncthreads();

  const int w = tid >> 5, lane = tid & 31, nl = lane & 15;
  f32x8 acc[4];
#pragma unroll
  for (int i = 0; i < 4; i++) {
    float bv = bias[w * 64 + i * 16 + nl];
#pragma unroll
    for (int e = 0; e < 8; e++) acc[i][e] = bv;
  }

  bf16x16 a = load_afrag(Ald, strideK, 0);
  bf16x16 bb[4];
#pragma unroll
  for (int i = 0; i < 4; i++) bb[i] = load_bfrag(WT, FOURH, 0, w * 64 + i * 16);

  for (int kc = 0; kc < KC; kc++) {
    bf16x16 an = a;
    bf16x16 bn[4] = {bb[0], bb[1], bb[2], bb[3]};
    if (kc + 1 < KC) {                        // prefetch next iteration
      an = load_afrag(Ald, strideK, kc + 1);
#pragma unroll
      for (int i = 0; i < 4; i++)
        bn[i] = load_bfrag(WT, FOURH, kc + 1, w * 64 + i * 16);
    }
#pragma unroll
    for (int i = 0; i < 4; i++) acc[i] = wmma_bf16(a, bb[i], acc[i]);
    a = an;
#pragma unroll
    for (int i = 0; i < 4; i++) bb[i] = bn[i];
  }

#pragma unroll
  for (int i = 0; i < 4; i++)
    *(f32x8*)(G + (((long)t * 32 + w * 4 + i) * 32 + lane) * 8) = acc[i];
}

// ---------------------------------------------------------------------------
// K2: LSTM recurrence over T steps.  Single persistent workgroup; recurrent
// weights live in VGPRs (16 B-frags/wave), h/c/gates in LDS.  Next step's
// 32KB pregate block is async-copied into a double-buffered LDS tile.
// ---------------------------------------------------------------------------
__global__ __launch_bounds__(256) void k_lstm_rec(
    const float* __restrict__ G, int T, const bf16_t* __restrict__ WhhT,
    float* __restrict__ hsOut, float* __restrict__ hFin, float* __restrict__ cFin) {
  __shared__ float gates[BT * FOURH];
  __shared__ float cS[BT * HD];
  __shared__ float hS[BT * HD];
  __shared__ __align__(64) bf16_t hA[BT * HD];
#if HAVE_ASYNC
  __shared__ __align__(128) float Gld[2][32 * 32 * 8];   // 2 x 32KB
#endif
  const int tid = threadIdx.x, w = tid >> 5, lane = tid & 31;
  const int nl = lane & 15, hi = lane >> 4;

  bf16x16 Bf[4][4];                       // resident Whh^T fragments
#pragma unroll
  for (int kc = 0; kc < 4; kc++)
#pragma unroll
    for (int i = 0; i < 4; i++)
      Bf[kc][i] = load_bfrag(WhhT, FOURH, kc, w * 64 + i * 16);

  for (int idx = tid; idx < BT * HD; idx += 256) {
    cS[idx] = 0.0f; hS[idx] = 0.0f; hA[idx] = f2bf(0.0f);
  }
#if HAVE_ASYNC
  {                                        // preload step 0 block
    const char* gsrc = (const char*)G + (size_t)tid * 128;
    char* ldst = (char*)&Gld[0][0] + (size_t)tid * 128;
#pragma unroll
    for (int k = 0; k < 8; k++) async_g2l_b128(gsrc + k * 16, ldst + k * 16);
  }
#endif
  __syncthreads();

  for (int t = 0; t < T; t++) {
#if HAVE_ASYNC
    if (t + 1 < T) {                       // start fetching step t+1
      const char* gsrc = (const char*)G + (size_t)(t + 1) * 32768 + (size_t)tid * 128;
      char* ldst = (char*)&Gld[(t + 1) & 1][0] + (size_t)tid * 128;
#pragma unroll
      for (int k = 0; k < 8; k++) async_g2l_b128(gsrc + k * 16, ldst + k * 16);
    }
    wait_async_le8();                      // step-t batch complete (this wave)
    __syncthreads();                       // ... and for all waves
#endif
    f32x8 acc[4];
#pragma unroll
    for (int i = 0; i < 4; i++) {
#if HAVE_ASYNC
      acc[i] = *(const f32x8*)(&Gld[t & 1][((w * 4 + i) * 32 + lane) * 8]);
#else
      acc[i] = *(const f32x8*)(G + (((long)t * 32 + w * 4 + i) * 32 + lane) * 8);
#endif
    }
#pragma unroll
    for (int kc = 0; kc < 4; kc++) {
      bf16x16 a = load_afrag(hA, HD, kc);
#pragma unroll
      for (int i = 0; i < 4; i++) acc[i] = wmma_bf16(a, Bf[kc][i], acc[i]);
    }
#pragma unroll
    for (int i = 0; i < 4; i++) {
      int nn = w * 64 + i * 16 + nl;
#pragma unroll
      for (int e = 0; e < 8; e++) gates[(e + 8 * hi) * FOURH + nn] = acc[i][e];
    }
    __syncthreads();
#pragma unroll
    for (int r = 0; r < 8; r++) {
      int p = tid + 256 * r;
      int b = p >> 7, j = p & 127;
      float gi = gates[b * FOURH + j];
      float gf = gates[b * FOURH + HD + j];
      float gg = gates[b * FOURH + 2 * HD + j];
      float go = gates[b * FOURH + 3 * HD + j];
      float c = sigm(gf) * cS[p] + sigm(gi) * tanh_(gg);
      float h = sigm(go) * tanh_(c);
      cS[p] = c; hS[p] = h; hA[p] = f2bf(h);
      if (hsOut) hsOut[((long)t * BT + b) * HD + j] = h;
    }
    __syncthreads();
    if (t + 2 < T)                         // warm L2 two steps ahead
      __builtin_prefetch(&G[((long)(t + 2) * BT) * FOURH + tid * 32], 0, 0);
  }
  for (int idx = tid; idx < BT * HD; idx += 256) {
    hFin[idx] = hS[idx]; cFin[idx] = cS[idx];
  }
}

// ---------------------------------------------------------------------------
// K3: autoregressive decoder, single persistent workgroup, 2047 steps.
// Layer gate matmuls use stacked [Wih | Whh] so each is one WMMA K-chain;
// B-fragment loads for chunk kc+1 are issued before the chunk-kc WMMAs.
// ---------------------------------------------------------------------------
__global__ __launch_bounds__(256) void k_decoder(
    const bf16_t* __restrict__ W0, const bf16_t* __restrict__ W1,
    const bf16_t* __restrict__ Wfc,
    const float* __restrict__ b0, const float* __restrict__ b1,
    const float* __restrict__ bfc,
    const float* __restrict__ h0i, const float* __restrict__ c0i,
    const float* __restrict__ h1i, const float* __restrict__ c1i,
    float* __restrict__ out, int T) {
  __shared__ float gates[BT * FOURH];
  __shared__ float c0S[BT * HD], c1S[BT * HD];
  __shared__ __align__(64) bf16_t A0[BT * 224];   // [inpt(96) | h0(128)]
  __shared__ __align__(64) bf16_t A1[BT * 256];   // [h0(128)  | h1(128)]
  const int tid = threadIdx.x, w = tid >> 5, lane = tid & 31;
  const int nl = lane & 15, hi = lane >> 4;

  for (int idx = tid; idx < BT * HD; idx += 256) {
    int b = idx >> 7, j = idx & 127;
    float h0 = h0i[idx], h1 = h1i[idx];
    c0S[idx] = c0i[idx]; c1S[idx] = c1i[idx];
    bf16_t h0b = f2bf(h0);
    A0[b * 224 + 96 + j] = h0b;
    A1[b * 256 + j] = h0b;
    A1[b * 256 + 128 + j] = f2bf(h1);
  }
  for (int idx = tid; idx < BT * 96; idx += 256) {
    int b = idx / 96, n = idx % 96;
    A0[b * 224 + n] = f2bf(0.0f);                 // inpt_0 = 0
    if (n < OUTD) out[(long)b * OUTD + n] = 0.0f; // outputs[0] = 0
  }
  __syncthreads();

  for (int t = 1; t < T; t++) {
    // ---- layer 0: gates = [inpt|h0] @ W0 + b0 (K = 224, 7 chunks) ----
    {
      f32x8 acc[4];
#pragma unroll
      for (int i = 0; i < 4; i++) {
        float bv = b0[w * 64 + i * 16 + nl];
#pragma unroll
        for (int e = 0; e < 8; e++) acc[i][e] = bv;
      }
      bf16x16 bb[4];
#pragma unroll
      for (int i = 0; i < 4; i++) bb[i] = load_bfrag(W0, FOURH, 0, w * 64 + i * 16);
#pragma unroll
      for (int kc = 0; kc < 7; kc++) {
        bf16x16 a = load_afrag(A0, 224, kc);
        bf16x16 bn[4] = {bb[0], bb[1], bb[2], bb[3]};
        if (kc + 1 < 7) {
#pragma unroll
          for (int i = 0; i < 4; i++)
            bn[i] = load_bfrag(W0, FOURH, kc + 1, w * 64 + i * 16);
        }
#pragma unroll
        for (int i = 0; i < 4; i++) acc[i] = wmma_bf16(a, bb[i], acc[i]);
#pragma unroll
        for (int i = 0; i < 4; i++) bb[i] = bn[i];
      }
#pragma unroll
      for (int i = 0; i < 4; i++) {
        int nn = w * 64 + i * 16 + nl;
#pragma unroll
        for (int e = 0; e < 8; e++) gates[(e + 8 * hi) * FOURH + nn] = acc[i][e];
      }
    }
    __syncthreads();
#pragma unroll
    for (int r = 0; r < 8; r++) {
      int p = tid + 256 * r;
      int b = p >> 7, j = p & 127;
      float gi = gates[b * FOURH + j];
      float gf = gates[b * FOURH + HD + j];
      float gg = gates[b * FOURH + 2 * HD + j];
      float go = gates[b * FOURH + 3 * HD + j];
      float c = sigm(gf) * c0S[p] + sigm(gi) * tanh_(gg);
      float h = sigm(go) * tanh_(c);
      c0S[p] = c;
      bf16_t hb = f2bf(h);
      A0[b * 224 + 96 + j] = hb;
      A1[b * 256 + j] = hb;
    }
    __syncthreads();
    // ---- layer 1: gates = [h0|h1] @ W1 + b1 (K = 256, 8 chunks) ----
    {
      f32x8 acc[4];
#pragma unroll
      for (int i = 0; i < 4; i++) {
        float bv = b1[w * 64 + i * 16 + nl];
#pragma unroll
        for (int e = 0; e < 8; e++) acc[i][e] = bv;
      }
      bf16x16 bb[4];
#pragma unroll
      for (int i = 0; i < 4; i++) bb[i] = load_bfrag(W1, FOURH, 0, w * 64 + i * 16);
#pragma unroll
      for (int kc = 0; kc < 8; kc++) {
        bf16x16 a = load_afrag(A1, 256, kc);
        bf16x16 bn[4] = {bb[0], bb[1], bb[2], bb[3]};
        if (kc + 1 < 8) {
#pragma unroll
          for (int i = 0; i < 4; i++)
            bn[i] = load_bfrag(W1, FOURH, kc + 1, w * 64 + i * 16);
        }
#pragma unroll
        for (int i = 0; i < 4; i++) acc[i] = wmma_bf16(a, bb[i], acc[i]);
#pragma unroll
        for (int i = 0; i < 4; i++) bb[i] = bn[i];
      }
#pragma unroll
      for (int i = 0; i < 4; i++) {
        int nn = w * 64 + i * 16 + nl;
#pragma unroll
        for (int e = 0; e < 8; e++) gates[(e + 8 * hi) * FOURH + nn] = acc[i][e];
      }
    }
    __syncthreads();
#pragma unroll
    for (int r = 0; r < 8; r++) {
      int p = tid + 256 * r;
      int b = p >> 7, j = p & 127;
      float gi = gates[b * FOURH + j];
      float gf = gates[b * FOURH + HD + j];
      float gg = gates[b * FOURH + 2 * HD + j];
      float go = gates[b * FOURH + 3 * HD + j];
      float c = sigm(gf) * c1S[p] + sigm(gi) * tanh_(gg);
      float h = sigm(go) * tanh_(c);
      c1S[p] = c;
      A1[b * 256 + 128 + j] = f2bf(h);
    }
    __syncthreads();
    // ---- fc: out = sigmoid(h1 @ Wfc + bfc), N = 96 (6 tiles) ----
    if (w < 6) {
      f32x8 acc;
      float bv = bfc[w * 16 + nl];
#pragma unroll
      for (int e = 0; e < 8; e++) acc[e] = bv;
      bf16x16 bb = load_bfrag(Wfc, 96, 0, w * 16);
#pragma unroll
      for (int kc = 0; kc < 4; kc++) {
        bf16x16 a = load_afrag(A1 + 128, 256, kc);   // h1 half of A1
        bf16x16 bn = bb;
        if (kc + 1 < 4) bn = load_bfrag(Wfc, 96, kc + 1, w * 16);
        acc = wmma_bf16(a, bb, acc);
        bb = bn;
      }
      int n = w * 16 + nl;
#pragma unroll
      for (int e = 0; e < 8; e++) {
        int m = e + 8 * hi;
        float s = sigm(acc[e]);
        if (n < OUTD) {
          out[((long)t * BT + m) * OUTD + n] = s;
          A0[m * 224 + n] = f2bf(s);                 // next step's input
        }
      }
    }
    __syncthreads();
  }
}

// ---------------------------------------------------------------------------
extern "C" void kernel_launch(void* const* d_in, const int* in_sizes, int n_in,
                              void* d_out, int out_size, void* d_ws, size_t ws_size,
                              hipStream_t stream) {
  (void)in_sizes; (void)n_in; (void)out_size; (void)ws_size;
  const float* src   = (const float*)d_in[0];
  const float* eWih0 = (const float*)d_in[2];
  const float* eWhh0 = (const float*)d_in[3];
  const float* ebih0 = (const float*)d_in[4];
  const float* ebhh0 = (const float*)d_in[5];
  const float* eWih1 = (const float*)d_in[6];
  const float* eWhh1 = (const float*)d_in[7];
  const float* ebih1 = (const float*)d_in[8];
  const float* ebhh1 = (const float*)d_in[9];
  const float* dWih0 = (const float*)d_in[10];
  const float* dWhh0 = (const float*)d_in[11];
  const float* dbih0 = (const float*)d_in[12];
  const float* dbhh0 = (const float*)d_in[13];
  const float* dWih1 = (const float*)d_in[14];
  const float* dWhh1 = (const float*)d_in[15];
  const float* dbih1 = (const float*)d_in[16];
  const float* dbhh1 = (const float*)d_in[17];
  const float* fcW   = (const float*)d_in[18];
  const float* fcb   = (const float*)d_in[19];

  char* ws = (char*)d_ws;
  size_t off = 0;
  auto alloc = [&](size_t bytes) -> void* {
    off = (off + 255) & ~(size_t)255;
    void* p = ws + off;
    off += bytes;
    return p;
  };
  bf16_t* WTih0 = (bf16_t*)alloc((size_t)KC0 * 32 * FOURH * 2);
  bf16_t* WThh0 = (bf16_t*)alloc((size_t)HD * FOURH * 2);
  bf16_t* WTih1 = (bf16_t*)alloc((size_t)HD * FOURH * 2);
  bf16_t* WThh1 = (bf16_t*)alloc((size_t)HD * FOURH * 2);
  bf16_t* WTd0  = (bf16_t*)alloc((size_t)224 * FOURH * 2);
  bf16_t* WTd1  = (bf16_t*)alloc((size_t)256 * FOURH * 2);
  bf16_t* WTfc  = (bf16_t*)alloc((size_t)HD * 96 * 2);
  float* be0 = (float*)alloc(FOURH * 4);
  float* be1 = (float*)alloc(FOURH * 4);
  float* bd0 = (float*)alloc(FOURH * 4);
  float* bd1 = (float*)alloc(FOURH * 4);
  float* bfc = (float*)alloc(96 * 4);
  float* h0F = (float*)alloc(BT * HD * 4);
  float* c0F = (float*)alloc(BT * HD * 4);
  float* h1F = (float*)alloc(BT * HD * 4);
  float* c1F = (float*)alloc(BT * HD * 4);
  float* hs0 = (float*)alloc((size_t)SRCLEN * BT * HD * 4);
  float* G   = (float*)alloc((size_t)SRCLEN * BT * FOURH * 4);  // reused L0/L1

  k_prep<<<256, 256, 0, stream>>>(
      eWih0, eWhh0, ebih0, ebhh0, eWih1, eWhh1, ebih1, ebhh1,
      dWih0, dWhh0, dbih0, dbhh0, dWih1, dWhh1, dbih1, dbhh1, fcW, fcb,
      WTih0, WThh0, WTih1, WThh1, WTd0, WTd1, WTfc,
      be0, be1, bd0, bd1, bfc);

  // Encoder layer 0: pregates from src [B][T][IN], row (t,m) = src[m][t][:]
  k_pregate<<<SRCLEN, 256, 0, stream>>>(
      src, (long)SRCLEN * IND, (long)IND, IND, KC0, WTih0, be0, G);
  k_lstm_rec<<<1, 256, 0, stream>>>(G, SRCLEN, WThh0, hs0, h0F, c0F);

  // Encoder layer 1: pregates from hs0 [T][B][H]
  k_pregate<<<SRCLEN, 256, 0, stream>>>(
      hs0, (long)HD, (long)BT * HD, HD, 4, WTih1, be1, G);
  k_lstm_rec<<<1, 256, 0, stream>>>(G, SRCLEN, WThh1, nullptr, h1F, c1F);

  // Decoder (autoregressive, writes d_out incl. zeroed t=0 row)
  k_decoder<<<1, 256, 0, stream>>>(
      WTd0, WTd1, WTfc, bd0, bd1, bfc, h0F, c0F, h1F, c1F,
      (float*)d_out, TRGLEN);
}
